// SlidingTransformer_29738353557711
// MI455X (gfx1250) — matrix operand order repaired and checked by reading, not compiled, not run on hardware
//
#include <hip/hip_runtime.h>
#include <hip/hip_bf16.h>

// ---------------- problem constants ----------------
constexpr int NB    = 8;      // batch
constexpr int NQ    = 256;    // query tokens
constexpr int ND    = 1024;   // model dim
constexpr int NH    = 16;     // heads
constexpr int NHD   = 64;     // head dim
constexpr int NDFF  = 4096;   // ffn dim
constexpr int NLY   = 4;      // layers
constexpr int NCACHE= 1024;
constexpr int NKLEN = 1280;   // CACHE + Q
constexpr int MQ    = NB * NQ;
constexpr int NTOK  = 4;      // tokens per timestep (block-causal granularity)

// ---------------- WMMA types / helpers ----------------
typedef __attribute__((ext_vector_type(16))) __bf16 v16bf;
typedef __attribute__((ext_vector_type(8)))  float  v8f;
typedef __attribute__((ext_vector_type(2)))  float  vf2;
typedef __attribute__((ext_vector_type(2)))  __bf16 vbf2;

__device__ __forceinline__ unsigned short f2bfbits(float f) {
    union { float f; unsigned u; } x; x.f = f;
    unsigned r = (x.u + 0x7FFFu + ((x.u >> 16) & 1u)) >> 16;   // RNE
    return (unsigned short)r;
}
// hot path: packed conversion; backend may select v_cvt_pk_bf16_f32
__device__ __forceinline__ unsigned pack2(float lo, float hi) {
    vf2 f; f.x = lo; f.y = hi;
    vbf2 b = __builtin_convertvector(f, vbf2);
    union { vbf2 b; unsigned u; } u; u.b = b; return u.u;
}
__device__ __forceinline__ v8f v8zero() {
    v8f z; for (int i = 0; i < 8; ++i) z[i] = 0.0f; return z;
}
__device__ __forceinline__ v8f wmma_bf16(v16bf a, v16bf b, v8f c) {
    return __builtin_amdgcn_wmma_f32_16x16x32_bf16(
        false, a, false, b, (short)0, c, false, false);
}
union Frag128 { uint4 q[2]; v16bf v; };

// ---------------- LayerNorm (one block per row, D=1024) ----------------
__global__ __launch_bounds__(256) void ln_kernel(
    const float* __restrict__ in, const float* __restrict__ g,
    const float* __restrict__ b, float* __restrict__ out,
    long long inStride, long long outStride)
{
    const float* row  = in  + (long long)blockIdx.x * inStride;
    float*       orow = out + (long long)blockIdx.x * outStride;
    const int tid = threadIdx.x;

    float v[4], s = 0.f, ss = 0.f;
    for (int i = 0; i < 4; ++i) {
        v[i] = row[tid + 256 * i];
        s += v[i]; ss += v[i] * v[i];
    }
    for (int m = 16; m >= 1; m >>= 1) {
        s  += __shfl_xor(s,  m, 32);
        ss += __shfl_xor(ss, m, 32);
    }
    __shared__ float s1[8], s2[8];
    if ((tid & 31) == 0) { s1[tid >> 5] = s; s2[tid >> 5] = ss; }
    __syncthreads();
    float ts = 0.f, tss = 0.f;
    for (int i = 0; i < 8; ++i) { ts += s1[i]; tss += s2[i]; }
    const float mean = ts * (1.0f / ND);
    const float var  = tss * (1.0f / ND) - mean * mean;
    const float rstd = rsqrtf(var + 1e-5f);
    for (int i = 0; i < 4; ++i) {
        int c = tid + 256 * i;
        orow[c] = (v[i] - mean) * rstd * g[c] + b[c];
    }
}

// ---------------- WMMA GEMM: C = epi(A @ W + bias [+ R]) ----------------
constexpr int BM = 128, BN = 128, BK = 32;
constexpr int EPI_BIAS = 0, EPI_RES = 1, EPI_GELU = 2;

struct TileRegs { float4 a[2][2]; float b[2][8]; };

__device__ __forceinline__ void load_tile(
    const float* __restrict__ A, const float* __restrict__ W,
    int bm0, int bn0, int k0, int K, int N, int tid, TileRegs& t)
{
    for (int i = 0; i < 2; ++i) {
        int cc = tid + 256 * i;
        int r = cc >> 2, c8 = (cc & 3) << 3;
        const float4* p = reinterpret_cast<const float4*>(
            A + (size_t)(bm0 + r) * K + (k0 + c8));
        t.a[i][0] = p[0];
        t.a[i][1] = p[1];
    }
    for (int i = 0; i < 2; ++i) {
        int cc = tid + 256 * i;
        int col = cc & 127, r8 = (cc >> 7) << 3;
        const float* p = W + (size_t)(k0 + r8) * N + (bn0 + col);
        for (int r = 0; r < 8; ++r) t.b[i][r] = p[(size_t)r * N];
    }
}

__device__ __forceinline__ void store_tile(
    unsigned short* __restrict__ As, unsigned short* __restrict__ Bt,
    int tid, const TileRegs& t)
{
    for (int i = 0; i < 2; ++i) {
        int cc = tid + 256 * i;
        int r = cc >> 2, c8 = (cc & 3) << 3;
        uint4 u;
        u.x = pack2(t.a[i][0].x, t.a[i][0].y);
        u.y = pack2(t.a[i][0].z, t.a[i][0].w);
        u.z = pack2(t.a[i][1].x, t.a[i][1].y);
        u.w = pack2(t.a[i][1].z, t.a[i][1].w);
        *reinterpret_cast<uint4*>(&As[r * BK + c8]) = u;     // ds_store_b128
    }
    for (int i = 0; i < 2; ++i) {
        int cc = tid + 256 * i;
        int col = cc & 127, r8 = (cc >> 7) << 3;
        uint4 u;
        u.x = pack2(t.b[i][0], t.b[i][1]);
        u.y = pack2(t.b[i][2], t.b[i][3]);
        u.z = pack2(t.b[i][4], t.b[i][5]);
        u.w = pack2(t.b[i][6], t.b[i][7]);
        *reinterpret_cast<uint4*>(&Bt[col * BK + r8]) = u;   // ds_store_b128
    }
}

template <int EPI>
__global__ __launch_bounds__(256) void gemm_bf16_kernel(
    const float* __restrict__ A, const float* __restrict__ W,
    const float* __restrict__ bias, const float* __restrict__ R,
    float* __restrict__ C, int M, int N, int K)
{
    __shared__ __attribute__((aligned(16))) unsigned short As[BM * BK]; // [row][k]
    __shared__ __attribute__((aligned(16))) unsigned short Bt[BN * BK]; // [col][k]

    const int tid  = threadIdx.x;
    const int lane = tid & 31, wave = tid >> 5;   // wave32
    const int wm = wave & 1, wn = wave >> 1;      // 2 (M) x 4 (N) waves
    const int half = lane >> 4, l16 = lane & 15;
    const int bm0 = blockIdx.y * BM, bn0 = blockIdx.x * BN;

    v8f acc[4][2];
    for (int mt = 0; mt < 4; ++mt)
        for (int nt = 0; nt < 2; ++nt) acc[mt][nt] = v8zero();

    TileRegs regs;
    load_tile(A, W, bm0, bn0, 0, K, N, tid, regs);

    for (int k0 = 0; k0 < K; k0 += BK) {
        store_tile(As, Bt, tid, regs);
        __syncthreads();
        if (k0 + BK < K) {
            __builtin_prefetch(&A[(size_t)(bm0 + (tid >> 1)) * K + k0 + BK], 0, 1);
            load_tile(A, W, bm0, bn0, k0 + BK, K, N, tid, regs); // overlap with WMMA
        }

        v16bf afrag[4], bfrag[2];
        for (int mt = 0; mt < 4; ++mt) {
            int r = wm * 64 + mt * 16 + l16;
            Frag128 u;
            u.q[0] = *reinterpret_cast<const uint4*>(&As[r * BK + 8 * half]);
            u.q[1] = *reinterpret_cast<const uint4*>(&As[r * BK + 16 + 8 * half]);
            afrag[mt] = u.v;
        }
        for (int nt = 0; nt < 2; ++nt) {
            int c = wn * 32 + nt * 16 + l16;
            Frag128 u;
            u.q[0] = *reinterpret_cast<const uint4*>(&Bt[c * BK + 8 * half]);
            u.q[1] = *reinterpret_cast<const uint4*>(&Bt[c * BK + 16 + 8 * half]);
            bfrag[nt] = u.v;
        }
        for (int mt = 0; mt < 4; ++mt)
            for (int nt = 0; nt < 2; ++nt)
                acc[mt][nt] = wmma_bf16(afrag[mt], bfrag[nt], acc[mt][nt]);
        __syncthreads();
    }

    for (int mt = 0; mt < 4; ++mt)
        for (int nt = 0; nt < 2; ++nt)
            for (int j = 0; j < 8; ++j) {
                int row = bm0 + wm * 64 + mt * 16 + j + 8 * half;
                int col = bn0 + wn * 32 + nt * 16 + l16;
                float v = acc[mt][nt][j] + bias[col];
                if (EPI == EPI_RES)  v += R[(size_t)row * N + col];
                if (EPI == EPI_GELU) v = 0.5f * v * (1.0f + erff(v * 0.70710678118654752f));
                C[(size_t)row * N + col] = v;
            }
}

// ---------------- K pack: cache+new -> bf16 [B,H,KLEN,HD] ----------------
__global__ __launch_bounds__(256) void pack_k_kernel(
    const float* __restrict__ pk, const float* __restrict__ knew,
    unsigned short* __restrict__ kout, int layer)
{
    size_t idx = (size_t)blockIdx.x * 256 + threadIdx.x;
    int hd = (int)(idx & (NHD - 1));
    size_t t = idx >> 6;
    int j  = (int)(t % NKLEN); t /= NKLEN;
    int hh = (int)(t % NH);
    int bb = (int)(t / NH);
    float kv;
    if (j < NCACHE)
        kv = pk[((((size_t)bb * NLY + layer) * NH + hh) * NCACHE + j) * NHD + hd];
    else
        kv = knew[((size_t)bb * NQ + (j - NCACHE)) * ND + hh * NHD + hd];
    kout[(((size_t)bb * NH + hh) * NKLEN + j) * NHD + hd] = f2bfbits(kv);
}

// ---------------- V pack: cache+new -> bf16 [B,H,HD,KLEN] (transposed) ----------------
__global__ __launch_bounds__(256) void pack_v_kernel(
    const float* __restrict__ pv, const float* __restrict__ vnew,
    unsigned short* __restrict__ vout, int layer)
{
    __shared__ unsigned short tile[64 * 64];   // [key][hd]
    const int jb = blockIdx.x * 64, h = blockIdx.y, b = blockIdx.z;
    const int tid = threadIdx.x;
    for (int i = 0; i < 16; ++i) {
        int idx = tid + 256 * i;
        int r = idx >> 6, c = idx & 63;
        int j = jb + r;
        float v;
        if (j < NCACHE)
            v = pv[((((size_t)b * NLY + layer) * NH + h) * NCACHE + j) * NHD + c];
        else
            v = vnew[((size_t)b * NQ + (j - NCACHE)) * ND + h * NHD + c];
        tile[r * 64 + c] = f2bfbits(v);
    }
    __syncthreads();
    unsigned short* vb = vout + ((size_t)b * NH + h) * (size_t)NHD * NKLEN;
    for (int i = 0; i < 16; ++i) {
        int idx = tid + 256 * i;
        int hd = idx >> 6, j = idx & 63;
        vb[(size_t)hd * NKLEN + jb + j] = tile[j * 64 + hd];
    }
}

// ---------------- flash-style attention ----------------
// grid = (Q/64, H, B), block = 128 (4 waves, 16 query rows each).
__global__ __launch_bounds__(128) void attn_kernel(
    const float* __restrict__ qbuf,             // [B,Q,D] f32
    const unsigned short* __restrict__ kbf,     // [B,H,KLEN,HD] bf16
    const unsigned short* __restrict__ vbf,     // [B,H,HD,KLEN] bf16 (transposed)
    float* __restrict__ ao)                     // [B,Q,D] f32
{
    __shared__ __attribute__((aligned(16))) unsigned short Vs[64 * 64];    // [hd][key]
    __shared__ __attribute__((aligned(16))) unsigned short Ps[4][16 * 64]; // [wave][q][key]

    const int tid  = threadIdx.x;
    const int lane = tid & 31, wave = tid >> 5;
    const int half = lane >> 4, l16 = lane & 15;
    const int qt = blockIdx.x, h = blockIdx.y, b = blockIdx.z;
    const int qwbase = qt * 64 + wave * 16;

    v16bf qf[2];
    {
        const float* qp = qbuf + ((size_t)b * NQ + (qwbase + l16)) * ND + h * NHD;
        for (int kc = 0; kc < 2; ++kc) {
            int base = kc * 32 + 8 * half;
            float4 f0 = *reinterpret_cast<const float4*>(qp + base);
            float4 f1 = *reinterpret_cast<const float4*>(qp + base + 4);
            float4 f2 = *reinterpret_cast<const float4*>(qp + base + 16);
            float4 f3 = *reinterpret_cast<const float4*>(qp + base + 20);
            Frag128 u;
            u.q[0].x = pack2(f0.x * 0.125f, f0.y * 0.125f);
            u.q[0].y = pack2(f0.z * 0.125f, f0.w * 0.125f);
            u.q[0].z = pack2(f1.x * 0.125f, f1.y * 0.125f);
            u.q[0].w = pack2(f1.z * 0.125f, f1.w * 0.125f);
            u.q[1].x = pack2(f2.x * 0.125f, f2.y * 0.125f);
            u.q[1].y = pack2(f2.z * 0.125f, f2.w * 0.125f);
            u.q[1].z = pack2(f3.x * 0.125f, f3.y * 0.125f);
            u.q[1].w = pack2(f3.z * 0.125f, f3.w * 0.125f);
            qf[kc] = u.v;
        }
    }

    float mrow[8], lrow[8];
    v8f Oacc[4];
    for (int j = 0; j < 8; ++j) { mrow[j] = -1e30f; lrow[j] = 0.0f; }
    for (int nt = 0; nt < 4; ++nt) Oacc[nt] = v8zero();

    const unsigned short* kbase = kbf + ((size_t)b * NH + h) * (size_t)NKLEN * NHD;
    const unsigned short* vbase = vbf + ((size_t)b * NH + h) * (size_t)NHD * NKLEN;

    // only keys < NCACHE + (qt+1)*64 can be visible to this q-tile (uniform bound)
    const int klen_eff = NCACHE + (qt + 1) * 64;

    for (int kb = 0; kb < klen_eff; kb += 64) {
        __syncthreads();
        // stage V block [hd][64 keys] as contiguous b128 copies
        for (int i = 0; i < 4; ++i) {
            int cc = tid + 128 * i;
            int hd = cc >> 3, k8 = (cc & 7) << 3;
            uint4 d = *reinterpret_cast<const uint4*>(&vbase[(size_t)hd * NKLEN + kb + k8]);
            *reinterpret_cast<uint4*>(&Vs[hd * 64 + k8]) = d;
        }

        // S = (Q * scale) @ K^T over this 64-key block
        v8f S[4];
        for (int nt = 0; nt < 4; ++nt) S[nt] = v8zero();
        for (int nt = 0; nt < 4; ++nt) {
            int key = kb + nt * 16 + l16;
            for (int kc = 0; kc < 2; ++kc) {
                const unsigned short* kp = kbase + (size_t)key * NHD + kc * 32 + 8 * half;
                Frag128 u;
                u.q[0] = *reinterpret_cast<const uint4*>(kp);
                u.q[1] = *reinterpret_cast<const uint4*>(kp + 16);
                S[nt] = wmma_bf16(qf[kc], u.v, S[nt]);
            }
        }

        // per-row max; apply block-causal mask only in the non-cache tail
        float rm[8];
        for (int j = 0; j < 8; ++j) rm[j] = -1e30f;
        if (kb >= NCACHE) {           // uniform branch (kb, NCACHE block-aligned)
            for (int nt = 0; nt < 4; ++nt) {
                int kts = (kb + nt * 16 + l16 - NCACHE) / NTOK;   // key timestep
                for (int j = 0; j < 8; ++j) {
                    int qts = (qt * 64 + wave * 16 + j + 8 * half) / NTOK;
                    float sv = (kts > qts) ? -1e30f : S[nt][j];
                    S[nt][j] = sv;
                    rm[j] = fmaxf(rm[j], sv);
                }
            }
        } else {
            for (int nt = 0; nt < 4; ++nt)
                for (int j = 0; j < 8; ++j)
                    rm[j] = fmaxf(rm[j], S[nt][j]);
        }
        for (int j = 0; j < 8; ++j)
            for (int m = 1; m <= 8; m <<= 1)
                rm[j] = fmaxf(rm[j], __shfl_xor(rm[j], m, 32));

        float fac[8], rs[8];
        for (int j = 0; j < 8; ++j) {
            float mn = fmaxf(mrow[j], rm[j]);
            fac[j] = expf(mrow[j] - mn);
            mrow[j] = mn;
            rs[j] = 0.0f;
        }
        for (int nt = 0; nt < 4; ++nt)
            for (int j = 0; j < 8; ++j) {
                float p = expf(S[nt][j] - mrow[j]);
                rs[j] += p;
                Ps[wave][(j + 8 * half) * 64 + nt * 16 + l16] = f2bfbits(p);
            }
        for (int j = 0; j < 8; ++j)
            for (int m = 1; m <= 8; m <<= 1)
                rs[j] += __shfl_xor(rs[j], m, 32);
        for (int j = 0; j < 8; ++j) lrow[j] = lrow[j] * fac[j] + rs[j];
        for (int nt = 0; nt < 4; ++nt)
            for (int j = 0; j < 8; ++j) Oacc[nt][j] *= fac[j];

        __syncthreads();

        // O += P @ V
        for (int kc = 0; kc < 2; ++kc) {
            const unsigned short* pp = &Ps[wave][l16 * 64 + kc * 32 + 8 * half];
            Frag128 pu;
            pu.q[0] = *reinterpret_cast<const uint4*>(pp);
            pu.q[1] = *reinterpret_cast<const uint4*>(pp + 16);
            for (int nt = 0; nt < 4; ++nt) {
                int hd = nt * 16 + l16;
                const unsigned short* vp = &Vs[hd * 64 + kc * 32 + 8 * half];
                Frag128 vu;
                vu.q[0] = *reinterpret_cast<const uint4*>(vp);
                vu.q[1] = *reinterpret_cast<const uint4*>(vp + 16);
                Oacc[nt] = wmma_bf16(pu.v, vu.v, Oacc[nt]);
            }
        }
    }

    for (int nt = 0; nt < 4; ++nt)
        for (int j = 0; j < 8; ++j) {
            int qg = qt * 64 + wave * 16 + j + 8 * half;
            int hd = nt * 16 + l16;
            ao[((size_t)b * NQ + qg) * ND + h * NHD + hd] = Oacc[nt][j] / lrow[j];
        }
}

// ---------------- orchestration ----------------
extern "C" void kernel_launch(void* const* d_in, const int* in_sizes, int n_in,
                              void* d_out, int out_size, void* d_ws, size_t ws_size,
                              hipStream_t stream)
{
    const float* x_in = (const float*)d_in[0];
    const float* pk   = (const float*)d_in[1];
    const float* pv   = (const float*)d_in[2];
    /* d_in[3] = mask (recomputed analytically) */
    const float* Wq = (const float*)d_in[4];  const float* bq  = (const float*)d_in[5];
    const float* Wk = (const float*)d_in[6];  const float* bk  = (const float*)d_in[7];
    const float* Wv = (const float*)d_in[8];  const float* bv  = (const float*)d_in[9];
    const float* Wo = (const float*)d_in[10]; const float* bo  = (const float*)d_in[11];
    const float* g1 = (const float*)d_in[12]; const float* be1 = (const float*)d_in[13];
    const float* g2 = (const float*)d_in[14]; const float* be2 = (const float*)d_in[15];
    const float* W1 = (const float*)d_in[16]; const float* bf1 = (const float*)d_in[17];
    const float* W2 = (const float*)d_in[18]; const float* bf2 = (const float*)d_in[19];
    const float* gf = (const float*)d_in[20]; const float* bef = (const float*)d_in[21];
    float* out = (float*)d_out;

    char* w = (char*)d_ws;
    auto carve = [&](size_t bytes) {
        void* p = (void*)w;
        w += (bytes + 255) & ~(size_t)255;
        return p;
    };
    float* xw   = (float*)carve((size_t)MQ * ND * 4);
    float* xnw  = (float*)carve((size_t)MQ * ND * 4);
    float* qbw  = (float*)carve((size_t)MQ * ND * 4);
    float* kbw  = (float*)carve((size_t)MQ * ND * 4);
    float* vbw  = (float*)carve((size_t)MQ * ND * 4);
    float* aow  = (float*)carve((size_t)MQ * ND * 4);
    float* ffw  = (float*)carve((size_t)MQ * NDFF * 4);
    unsigned short* kbfw = (unsigned short*)carve((size_t)NB * NH * NKLEN * NHD * 2);
    unsigned short* vbfw = (unsigned short*)carve((size_t)NB * NH * NKLEN * NHD * 2);

    hipMemcpyAsync(xw, x_in, (size_t)MQ * ND * sizeof(float),
                   hipMemcpyDeviceToDevice, stream);

    const dim3 gD (ND   / BN, MQ / BM);
    const dim3 gF1(NDFF / BN, MQ / BM);
    const int  kBlocks = (NB * NH * NKLEN * NHD) / 256;

    for (int i = 0; i < NLY; ++i) {
        ln_kernel<<<MQ, 256, 0, stream>>>(xw, g1 + (size_t)i * ND, be1 + (size_t)i * ND,
                                          xnw, ND, ND);
        gemm_bf16_kernel<EPI_BIAS><<<gD, 256, 0, stream>>>(
            xnw, Wq + (size_t)i * ND * ND, bq + (size_t)i * ND, nullptr, qbw, MQ, ND, ND);
        gemm_bf16_kernel<EPI_BIAS><<<gD, 256, 0, stream>>>(
            xnw, Wk + (size_t)i * ND * ND, bk + (size_t)i * ND, nullptr, kbw, MQ, ND, ND);
        gemm_bf16_kernel<EPI_BIAS><<<gD, 256, 0, stream>>>(
            xnw, Wv + (size_t)i * ND * ND, bv + (size_t)i * ND, nullptr, vbw, MQ, ND, ND);
        pack_k_kernel<<<kBlocks, 256, 0, stream>>>(pk, kbw, kbfw, i);
        pack_v_kernel<<<dim3(NKLEN / 64, NH, NB), 256, 0, stream>>>(pv, vbw, vbfw, i);
        attn_kernel<<<dim3(NQ / 64, NH, NB), 128, 0, stream>>>(qbw, kbfw, vbfw, aow);
        gemm_bf16_kernel<EPI_RES><<<gD, 256, 0, stream>>>(
            aow, Wo + (size_t)i * ND * ND, bo + (size_t)i * ND, xw, xw, MQ, ND, ND);
        ln_kernel<<<MQ, 256, 0, stream>>>(xw, g2 + (size_t)i * ND, be2 + (size_t)i * ND,
                                          xnw, ND, ND);
        gemm_bf16_kernel<EPI_GELU><<<gF1, 256, 0, stream>>>(
            xnw, W1 + (size_t)i * ND * NDFF, bf1 + (size_t)i * NDFF, nullptr,
            ffw, MQ, NDFF, ND);
        gemm_bf16_kernel<EPI_RES><<<gD, 256, 0, stream>>>(
            ffw, W2 + (size_t)i * NDFF * ND, bf2 + (size_t)i * ND, xw, xw, MQ, ND, NDFF);
    }
    ln_kernel<<<NB, 256, 0, stream>>>(xw, gf, bef, out,
                                      (long long)NQ * ND, (long long)ND);
}
